// GRUModel_36086315221593
// MI455X (gfx1250) — compile-verified, matrix operand order checked by reading, not run on hardware
//
#include <hip/hip_runtime.h>
#include <hip/hip_bf16.h>
#include <stdint.h>

// ---------------------------------------------------------------------------
// GRU (H=768, bs=32, T=512) for MI455X / gfx1250, wave32.
// Phase 1: xg = emb @ W_ih^T + b_ih via v_wmma_f32_16x16x32_f16.
//          B tiles staged in LDS with double-buffered ASYNC global->LDS DMA
//          (global_load_async_to_lds_b128 + s_wait_asynccnt), A fragments
//          register-pipelined one k-step ahead.
// Phase 2: strictly serial GRU chain (hidden state carries ACROSS samples in
//          the reference scan) -> persistent 16-block kernel. Each block's
//          144x768 f16 W_hh slice (221 KB) is async-DMA'd into its WGP's
//          320 KB LDS ONCE, then ~8K steps run entirely out of LDS.
//          Double-buffered h in L2 + one atomic barrier per step.
// ---------------------------------------------------------------------------

typedef _Float16 v16h __attribute__((ext_vector_type(16)));
typedef _Float16 v8h  __attribute__((ext_vector_type(8)));
typedef float    v8f  __attribute__((ext_vector_type(8)));

#define H    768
#define H3   2304
#define BS   32
#define TT   512
#define MROWS (BS * TT)          // 16384

// phase-2 dynamic LDS: W slice + h + hg
#define REC_W_BYTES  ((size_t)144 * H * 2)          // 221184
#define REC_LDS_BYTES (REC_W_BYTES + (H + 144) * 4) // 224832  (< 320 KB/WGP)

// workspace layout (bytes); total ~183 MB
#define WS_BAR   ((size_t)0)                       // 2 x u32 barrier
#define WS_L     ((size_t)128)                     // 32 x int lengths
#define WS_H0    ((size_t)1024)                    // 768 f32 h buffer A
#define WS_H1    ((size_t)4096)                    // 768 f32 h buffer B
#define WS_WIH   ((size_t)8192)                    // 2304*768 f16
#define WS_WHH   (WS_WIH + (size_t)H3 * H * 2)     // 2304*768 f16
#define WS_EMB   (WS_WHH + (size_t)H3 * H * 2)     // 16384*768 f16
#define WS_XG    (WS_EMB + (size_t)MROWS * H * 2)  // 16384*2304 f32

// -------------------- CDNA5 async-copy helpers -----------------------------
__device__ __forceinline__ void async_copy_b128_to_lds(unsigned lds_off,
                                                       const void* gaddr) {
  // GV mode: VDST = LDS byte offset VGPR, VADDR = 64-bit global address.
  asm volatile("global_load_async_to_lds_b128 %0, %1, off"
               :: "v"(lds_off), "v"(gaddr) : "memory");
}

__device__ __forceinline__ void wait_async0() {
#if __has_builtin(__builtin_amdgcn_s_wait_asynccnt)
  __builtin_amdgcn_s_wait_asynccnt(0);
#else
  asm volatile("s_wait_asynccnt 0x0" ::: "memory");
#endif
}

// -------------------- init: lengths, h0, barrier ---------------------------
__global__ void gru_init(const int* __restrict__ mask, const float* __restrict__ gc,
                         int* __restrict__ Lv, float* __restrict__ hb0,
                         unsigned* __restrict__ bar) {
  int tid = threadIdx.x;
  if (tid < H) hb0[tid] = gc[tid];
  if (tid < BS) {
    int z = 0;
    for (int t = 0; t < TT; ++t) z += (mask[tid * TT + t] == 0);
    int L = TT - z;
    if (L < 1) L = 2;            // reference clamp
    Lv[tid] = L;
  }
  if (tid == 1000) { bar[0] = 0u; bar[1] = 0u; }
}

// -------------------- f32 -> f16 conversion --------------------------------
__global__ void cvt_f16(const float* __restrict__ in, _Float16* __restrict__ out, size_t n) {
  for (size_t i = (size_t)blockIdx.x * blockDim.x + threadIdx.x; i < n;
       i += (size_t)gridDim.x * blockDim.x)
    out[i] = (_Float16)in[i];
}

// -------------------- WMMA fragment helpers --------------------------------
// 16-bit A layout (ISA 7.12.2): lane<16 row M=lane holds K = k0+{0..7},k0+{16..23};
// lane>=16 row M=lane-16 holds K = k0+{8..15},k0+{24..31}. Both matrices here are
// [outer][K] row-major, so the per-lane fetch is two contiguous 16B chunks.
// B layout assumed mirrored (N across lanes, K within lane) — same fetch shape.
__device__ __forceinline__ v16h load_frag(const _Float16* p /* outer*K + k0 + half*8 */) {
  v8h lo = *(const v8h*)p;
  v8h hi = *(const v8h*)(p + 16);
  v16h f;
#pragma unroll
  for (int i = 0; i < 8; ++i) { f[i] = lo[i]; f[i + 8] = hi[i]; }
  return f;
}

__device__ __forceinline__ v8f wmma16(v16h a, v16h b, v8f c) {
  // (neg_a, A, neg_b, B, c_mod, C, reuse_a, reuse_b)
  return __builtin_amdgcn_wmma_f32_16x16x32_f16(false, a, false, b, (short)0, c,
                                                false, false);
}

// -------------------- phase 1: xg GEMM -------------------------------------
// grid: 4608 blocks x 256 thr. Block = (mblk 0..127 -> 128 M rows, nstrip
// 0..35 -> 64 N cols). The 8 waves share one 64x32 B tile per k-step, staged
// into double-buffered LDS by async DMA (1 b128 per thread per step). A
// fragments are loaded direct from global, pipelined one step ahead.
__global__ __launch_bounds__(256)
void gru_xg_gemm(const _Float16* __restrict__ A, const _Float16* __restrict__ W,
                 const float* __restrict__ bih, float* __restrict__ xg) {
  __shared__ __align__(64) _Float16 bsh[2][64 * 32];   // 2 x 4 KB

  const int tid  = threadIdx.x;
  const int lane = tid & 31;
  const int wv   = tid >> 5;
  const int mblk   = blockIdx.x & 127;   // 4608 = 36 * 128
  const int nstrip = blockIdx.x >> 7;
  const int m0 = mblk * 128 + wv * 16;
  const int n0 = nstrip * 64;
  const int r   = lane & 15;
  const int hlf = lane >> 4;

  // staging map: thread -> (row 0..63, 16B chunk 0..3) of the 64x32 B tile
  const int srow = tid >> 2, sch = tid & 3;
  const _Float16* gsrc = W + (size_t)(n0 + srow) * H + sch * 8;
  const unsigned lds0 = (unsigned)(uintptr_t)&bsh[0][srow * 32 + sch * 8];
  const unsigned lds1 = (unsigned)(uintptr_t)&bsh[1][srow * 32 + sch * 8];

  // prologue: fill buffer 0 with k0 = 0
  async_copy_b128_to_lds(lds0, gsrc);
  wait_async0();
  __syncthreads();

  v8f acc0 = {}, acc1 = {}, acc2 = {}, acc3 = {};
  const _Float16* arow = A + (size_t)(m0 + r) * H + hlf * 8;
  v16h a = load_frag(arow);

  for (int i = 0; i < 24; ++i) {                 // K = 768 = 24 * 32
    const int buf = i & 1;
    if (i + 1 < 24)                              // async-fill other buffer
      async_copy_b128_to_lds(buf ? lds0 : lds1, gsrc + (i + 1) * 32);

    v16h a_next = a;
    if (i + 1 < 24) a_next = load_frag(arow + (i + 1) * 32);  // pipelined A

    const _Float16* bb = &bsh[buf][0];
    acc0 = wmma16(a, load_frag(bb + (0 * 16 + r) * 32 + hlf * 8), acc0);
    acc1 = wmma16(a, load_frag(bb + (1 * 16 + r) * 32 + hlf * 8), acc1);
    acc2 = wmma16(a, load_frag(bb + (2 * 16 + r) * 32 + hlf * 8), acc2);
    acc3 = wmma16(a, load_frag(bb + (3 * 16 + r) * 32 + hlf * 8), acc3);

    wait_async0();          // next buffer ready (waits only this wave's DMA)
    __syncthreads();        // all waves done reading buf + DMA visible
    a = a_next;
  }

  // C/D layout: element e -> M = m0 + e + hlf*8, N = n0 + subtile*16 + (lane&15)
  float b0 = bih[n0 + 0 * 16 + r], b1 = bih[n0 + 1 * 16 + r];
  float b2 = bih[n0 + 2 * 16 + r], b3 = bih[n0 + 3 * 16 + r];
#pragma unroll
  for (int e = 0; e < 8; ++e) {
    size_t row = (size_t)(m0 + hlf * 8 + e) * H3;
    xg[row + n0 + 0 * 16 + r] = acc0[e] + b0;
    xg[row + n0 + 1 * 16 + r] = acc1[e] + b1;
    xg[row + n0 + 2 * 16 + r] = acc2[e] + b2;
    xg[row + n0 + 3 * 16 + r] = acc3[e] + b3;
  }
}

// -------------------- phase 2: serial GRU chain ----------------------------
// 16 persistent blocks x 288 thr (9 waves). Block s owns h-cols [48s,48s+48)
// => W_hh rows {g*768 + 48s + j} for g in {r,z,n}. The 221 KB f16 W slice is
// async-DMA'd into LDS once (WGP has 320 KB), then every step's matvec runs
// from LDS. Double-buffered h in L2, one atomic barrier per step.
__global__ __launch_bounds__(288)
void gru_rec(const float* __restrict__ xg, const _Float16* __restrict__ Whh,
             const float* __restrict__ bhh, const int* __restrict__ Lv,
             float* __restrict__ hb0, float* __restrict__ hb1,
             unsigned* __restrict__ bar, float* __restrict__ out) {
  extern __shared__ __align__(64) char smem[];
  _Float16* wsl  = (_Float16*)smem;                         // 144*768 f16
  float*    h_s  = (float*)(smem + REC_W_BYTES);            // 768 f32
  float*    hg_s = (float*)(smem + REC_W_BYTES + H * 4);    // 144 f32

  const int tid  = threadIdx.x;
  const int lane = tid & 31;
  const int wv   = tid >> 5;             // 0..8
  const int s    = blockIdx.x;           // slice 0..15
  volatile unsigned* vgen = bar + 1;
  int p = 0;
  float* hbuf[2] = {hb0, hb1};

  // one-time: DMA this block's W_hh slice into LDS (13824 b128s, 48/thread)
  for (int c = tid; c < 144 * 96; c += 288) {
    const int rl = c / 96, ch = c % 96;          // row-local, 16B chunk
    const int grow = (rl / 48) * H + s * 48 + (rl % 48);
    async_copy_b128_to_lds((unsigned)(uintptr_t)&wsl[rl * H + ch * 8],
                           Whh + (size_t)grow * H + ch * 8);
  }
  wait_async0();
  __syncthreads();

  for (int b = 0; b < BS; ++b) {
    const int Lb = Lv[b];                // uniform across blocks -> same barrier count
    for (int t = 0; t < Lb; ++t) {
      const float* cur = hbuf[p];
      float* nxt = hbuf[1 - p];
      for (int i = tid; i < H; i += 288) h_s[i] = cur[i];
      __syncthreads();

      // hoist this lane's 24-element h chunk into registers (one LDS read/step)
      float hreg[24];
#pragma unroll
      for (int c = 0; c < 24; ++c) hreg[c] = h_s[lane * 24 + c];

      // matvec: wave wv computes rows [16*wv, 16*wv+16), all from LDS
#pragma unroll 1
      for (int rr = 0; rr < 16; ++rr) {
        const int rl   = wv * 16 + rr;           // 0..143
        const _Float16* wp = wsl + rl * H + lane * 24;
        float pa = 0.f;
#pragma unroll
        for (int c = 0; c < 24; ++c) pa += (float)wp[c] * hreg[c];
#pragma unroll
        for (int off = 16; off > 0; off >>= 1) pa += __shfl_down(pa, off, 32);
        if (lane == 0)
          hg_s[rl] = pa + bhh[(rl / 48) * H + s * 48 + (rl % 48)];
      }
      __syncthreads();

      if (tid < 48) {
        const int col = s * 48 + tid;
        const size_t xb = ((size_t)b * TT + t) * H3 + col;  // xg already has b_ih
        if (t + 1 < Lb) __builtin_prefetch(&xg[xb + H3], 0, 1);  // next step's xg
        float xr = xg[xb], xz = xg[xb + H], xn = xg[xb + 2 * H];
        float rg = 1.f / (1.f + __expf(-(xr + hg_s[tid])));
        float zg = 1.f / (1.f + __expf(-(xz + hg_s[48 + tid])));
        float ng = tanhf(xn + rg * hg_s[96 + tid]);
        nxt[col] = (1.f - zg) * ng + zg * h_s[col];
      }
      __threadfence();
      __syncthreads();

      if (tid == 0) {                    // sense-reversing barrier over 16 blocks
        unsigned g = *vgen;
        if (atomicAdd(bar, 1u) == 15u) {
          bar[0] = 0u;
          __threadfence();
          atomicAdd(bar + 1, 1u);
        } else {
          while (*vgen == g) __builtin_amdgcn_s_sleep(1);
        }
      }
      __syncthreads();
      p ^= 1;                            // hbuf[p] now holds the fresh h
    }
    if (tid < 48) out[(size_t)b * H + s * 48 + tid] = hbuf[p][s * 48 + tid];
    __syncthreads();
  }
  // hF output = final hidden after sample 31
  if (tid < 48) out[(size_t)BS * H + s * 48 + tid] = hbuf[p][s * 48 + tid];
}

// -------------------- launcher ---------------------------------------------
extern "C" void kernel_launch(void* const* d_in, const int* in_sizes, int n_in,
                              void* d_out, int out_size, void* d_ws, size_t ws_size,
                              hipStream_t stream) {
  const float* emb  = (const float*)d_in[0];   // [32,512,768]
  const int*   mask = (const int*)d_in[1];     // [32,512]
  const float* gc   = (const float*)d_in[2];   // [1,1,768]
  const float* wih  = (const float*)d_in[3];   // [2304,768]
  const float* whh  = (const float*)d_in[4];   // [2304,768]
  const float* bih  = (const float*)d_in[5];   // [2304]
  const float* bhh  = (const float*)d_in[6];   // [2304]
  float* out = (float*)d_out;                  // 32*768 + 768

  char* ws = (char*)d_ws;
  unsigned*  bar   = (unsigned*)(ws + WS_BAR);
  int*       Lv    = (int*)(ws + WS_L);
  float*     hb0   = (float*)(ws + WS_H0);
  float*     hb1   = (float*)(ws + WS_H1);
  _Float16*  wih_h = (_Float16*)(ws + WS_WIH);
  _Float16*  whh_h = (_Float16*)(ws + WS_WHH);
  _Float16*  emb_h = (_Float16*)(ws + WS_EMB);
  float*     xg    = (float*)(ws + WS_XG);

  gru_init<<<1, 1024, 0, stream>>>(mask, gc, Lv, hb0, bar);

  cvt_f16<<<2048, 256, 0, stream>>>(emb, emb_h, (size_t)MROWS * H);
  cvt_f16<<<2048, 256, 0, stream>>>(wih, wih_h, (size_t)H3 * H);
  cvt_f16<<<2048, 256, 0, stream>>>(whh, whh_h, (size_t)H3 * H);

  gru_xg_gemm<<<4608, 256, 0, stream>>>(emb_h, wih_h, bih, xg);

  // 221 KB dynamic LDS for the persistent recurrence kernel (WGP allows 320 KB)
  hipFuncSetAttribute((const void*)gru_rec,
                      hipFuncAttributeMaxDynamicSharedMemorySize,
                      (int)REC_LDS_BYTES);
  gru_rec<<<16, 288, REC_LDS_BYTES, stream>>>(xg, whh_h, bhh, Lv, hb0, hb1, bar, out);
}
